// GumbelSelection_36283883716962
// MI455X (gfx1250) — matrix operand order repaired and checked by reading, not compiled, not run on hardware
//
#include <hip/hip_runtime.h>
#include <hip/hip_bf16.h>
#include <math.h>

typedef __attribute__((ext_vector_type(16))) _Float16 v16h;
typedef __attribute__((ext_vector_type(8)))  _Float16 v8h;
typedef __attribute__((ext_vector_type(8)))  float    v8f;
typedef __attribute__((ext_vector_type(4)))  float    v4f;

#define BD    256   // 8 wave32 per workgroup
#define MAXNN 384
#define DDIM  128
#define NOUT  256
#define KSEL  8
#define NSUB  4
#define NEGINF (-__builtin_inff())

// ---------------- prep kernels ----------------

__global__ void cvt_w1_kernel(const float* __restrict__ w1, _Float16* __restrict__ w1h, int n) {
    int i = blockIdx.x * blockDim.x + threadIdx.x;
    if (i < n) w1h[i] = (_Float16)w1[i];
}

__global__ void scan_counts_kernel(const int* __restrict__ counts, int* __restrict__ offs, int B) {
    if (blockIdx.x == 0 && threadIdx.x == 0) {
        int acc = 0;
        for (int g = 0; g < B; ++g) { offs[g] = acc; acc += counts[g]; }
    }
}

// ---------------- fused per-graph kernel ----------------

__global__ __launch_bounds__(BD) void gumbel_fused_kernel(
    const float*    __restrict__ node_repr,
    const _Float16* __restrict__ w1h,
    const float*    __restrict__ b1,
    const float*    __restrict__ w2,
    const float*    __restrict__ b2,
    const float*    __restrict__ gumbel,
    const int*      __restrict__ sub_ids,
    const int*      __restrict__ offs,
    const int*      __restrict__ counts,
    int*            __restrict__ out_idx,
    float*          __restrict__ out_samples)
{
    __shared__ float sMean[DDIM];
    __shared__ float sB1[NOUT];
    __shared__ float sW2[NOUT];
    __shared__ float sY[MAXNN];   // logits -> y (kept for softmax)
    __shared__ float sWk[MAXNN];  // mutable copy for top-k
    __shared__ float sRed[BD];
    __shared__ float sRV[BD];
    __shared__ int   sRI[BD];
    __shared__ int   sIdx[KSEL];

    const int g   = blockIdx.x;
    const int t   = threadIdx.x;
    const int c   = counts[g];
    const int off = offs[g];

    // ---- pass 1: segment mean over this graph's rows (coalesced) ----
    {
        const int d    = t & (DDIM - 1);
        const int half = t >> 7;           // 0/1
        float s = 0.f;
        for (int r = half; r < c; r += 2)
            s += node_repr[(size_t)(off + r) * DDIM + d];
        sRed[t] = s;
        if (t < NOUT) { sB1[t] = b1[t]; sW2[t] = w2[t]; }
        __syncthreads();
        if (t < DDIM) sMean[t] = (sRed[t] + sRed[t + DDIM]) / (float)c;
        __syncthreads();
    }

    // ---- pass 2: WMMA GEMM  h=relu((x+mean)@w1+b1), logits=h@w2+b2 ----
    const int lane = t & 31;
    const int wave = t >> 5;
    const int lrow = lane & 15;
    const int lhi  = lane >> 4;       // 0: lanes 0-15, 1: lanes 16-31
    const int mT   = c >> 4;          // c is 128 or 384 -> multiple of 16
    const float b2v = b2[0];

    #pragma unroll 1
    for (int mi = wave; mi < mT; mi += 8) {
        const int m0 = mi << 4;
        const float* rowp = node_repr + (size_t)(off + m0 + lrow) * DDIM;

        // A fragments for all 4 K-steps, reused across all 16 N-tiles (32 VGPRs)
        v16h afr[4];
        #pragma unroll
        for (int ks = 0; ks < 4; ++ks) {
            const int kb = ks * 32 + (lhi << 3);
            v4f r0 = *(const v4f*)(rowp + kb);
            v4f r1 = *(const v4f*)(rowp + kb + 4);
            v4f r2 = *(const v4f*)(rowp + kb + 16);
            v4f r3 = *(const v4f*)(rowp + kb + 20);
            v4f m0v = *(const v4f*)(sMean + kb);
            v4f m1v = *(const v4f*)(sMean + kb + 4);
            v4f m2v = *(const v4f*)(sMean + kb + 16);
            v4f m3v = *(const v4f*)(sMean + kb + 20);
            #pragma unroll
            for (int e = 0; e < 4; ++e) {
                afr[ks][e]      = (_Float16)(r0[e] + m0v[e]);
                afr[ks][4 + e]  = (_Float16)(r1[e] + m1v[e]);
                afr[ks][8 + e]  = (_Float16)(r2[e] + m2v[e]);
                afr[ks][12 + e] = (_Float16)(r3[e] + m3v[e]);
            }
        }

        float acc[8];
        #pragma unroll
        for (int v = 0; v < 8; ++v) acc[v] = 0.f;

        // B-matrix base for this lane (lane -> N column, elements -> K in table order)
        const _Float16* bbase = w1h + (size_t)(lhi << 3) * NOUT + lrow;

        #pragma unroll 1           // keep register pressure bounded: one B frag live
        for (int ni = 0; ni < 16; ++ni) {
            const int n0 = ni << 4;
            v8f dmat = {0.f, 0.f, 0.f, 0.f, 0.f, 0.f, 0.f, 0.f};
            #pragma unroll
            for (int ks = 0; ks < 4; ++ks) {
                const _Float16* bp = bbase + (size_t)(ks * 32) * NOUT + n0;
                v16h bfr;
                #pragma unroll
                for (int e = 0; e < 8; ++e) { bfr[e] = bp[(size_t)e * NOUT]; bfr[8 + e] = bp[(size_t)(16 + e) * NOUT]; }
                dmat = __builtin_amdgcn_wmma_f32_16x16x32_f16(
                    false, afr[ks], false, bfr, (short)0, dmat, false, false);
            }
            // fused epilogue: relu(+b1) then dot with w2 (D layout: lane->N, vgpr->M)
            const int n  = n0 + lrow;
            const float bn = sB1[n];
            const float wn = sW2[n];
            #pragma unroll
            for (int v = 0; v < 8; ++v) {
                float h = dmat[v] + bn;
                h = fmaxf(h, 0.f);
                acc[v] = fmaf(h, wn, acc[v]);
            }
        }
        // reduce over the 16 lanes holding one N-slice each
        #pragma unroll
        for (int o = 1; o < 16; o <<= 1) {
            #pragma unroll
            for (int v = 0; v < 8; ++v) acc[v] += __shfl_xor(acc[v], o, 32);
        }
        if (lrow == 0) {
            #pragma unroll
            for (int v = 0; v < 8; ++v)
                sY[m0 + (lhi << 3) + v] = acc[v] + b2v;
        }
    }
    __syncthreads();

    // ---- mask: probs = min(dense, mask) == logits with -inf at sub_ids-1 ----
    if (t < NSUB) {
        int sj = sub_ids[g * NSUB + t] - 1;
        if (sj >= 0 && sj < MAXNN) sY[sj] = NEGINF;
    }
    __syncthreads();

    // ---- y = probs + gumbel (TAU=1); pad -> -inf ----
    const float* gum = gumbel + (size_t)g * MAXNN;
    for (int j = t; j < MAXNN; j += BD) {
        float y = (j < c) ? (sY[j] + gum[j]) : NEGINF;
        sY[j]  = y;
        sWk[j] = y;
    }
    __syncthreads();

    // ---- softmax stats ----
    float lm = NEGINF;
    for (int j = t; j < MAXNN; j += BD) lm = fmaxf(lm, sY[j]);
    sRed[t] = lm;
    __syncthreads();
    for (int s = BD / 2; s > 0; s >>= 1) {
        if (t < s) sRed[t] = fmaxf(sRed[t], sRed[t + s]);
        __syncthreads();
    }
    const float Mx = sRed[0];
    __syncthreads();

    float ls = 0.f;
    for (int j = t; j < MAXNN; j += BD) ls += expf(sY[j] - Mx);
    sRed[t] = ls;
    __syncthreads();
    for (int s = BD / 2; s > 0; s >>= 1) {
        if (t < s) sRed[t] += sRed[t + s];
        __syncthreads();
    }
    const float invS = 1.f / sRed[0];
    __syncthreads();

    // ---- top-8 (descending, ties -> lower index, matches lax.top_k) ----
    for (int k = 0; k < KSEL; ++k) {
        float bv = NEGINF; int bi = MAXNN;
        for (int j = t; j < MAXNN; j += BD) {
            float vj = sWk[j];
            if (vj > bv || (vj == bv && j < bi)) { bv = vj; bi = j; }
        }
        sRV[t] = bv; sRI[t] = bi;
        __syncthreads();
        for (int s = BD / 2; s > 0; s >>= 1) {
            if (t < s) {
                float ov = sRV[t + s]; int oi = sRI[t + s];
                if (ov > sRV[t] || (ov == sRV[t] && oi < sRI[t])) { sRV[t] = ov; sRI[t] = oi; }
            }
            __syncthreads();
        }
        if (t == 0) { sIdx[k] = sRI[0]; sWk[sRI[0]] = NEGINF; }
        __syncthreads();
    }

    // ---- outputs: samples = (khot - soft) + soft ; idx ----
    for (int j = t; j < MAXNN; j += BD) {
        float soft = expf(sY[j] - Mx) * invS;
        float kh = 0.f;
        #pragma unroll
        for (int k = 0; k < KSEL; ++k) if (sIdx[k] == j) kh = 1.f;
        out_samples[(size_t)g * MAXNN + j] = (kh - soft) + soft;
    }
    if (t < KSEL) out_idx[g * KSEL + t] = sIdx[t];
}

// ---------------- host launcher ----------------

extern "C" void kernel_launch(void* const* d_in, const int* in_sizes, int n_in,
                              void* d_out, int out_size, void* d_ws, size_t ws_size,
                              hipStream_t stream) {
    const float* node_repr = (const float*)d_in[0];
    const float* w1        = (const float*)d_in[1];
    const float* b1        = (const float*)d_in[2];
    const float* w2        = (const float*)d_in[3];
    const float* b2        = (const float*)d_in[4];
    const float* gumbel    = (const float*)d_in[5];
    // d_in[6] node2graph, d_in[7] local_ids, d_in[9] sub2graph, d_in[11] num_sub: not needed
    const int* counts  = (const int*)d_in[8];
    const int* sub_ids = (const int*)d_in[10];

    const int B = in_sizes[8];

    // workspace: [w1 in f16: 64KB][graph start offsets: B ints]
    _Float16* w1h  = (_Float16*)d_ws;
    int*      offs = (int*)((char*)d_ws + (size_t)DDIM * NOUT * sizeof(_Float16));

    cvt_w1_kernel<<<(DDIM * NOUT + 255) / 256, 256, 0, stream>>>(w1, w1h, DDIM * NOUT);
    scan_counts_kernel<<<1, 32, 0, stream>>>(counts, offs, B);

    int*   out_idx     = (int*)d_out;                       // B*8 int32 (bit pattern)
    float* out_samples = (float*)d_out + (size_t)B * KSEL;  // B*384 f32

    gumbel_fused_kernel<<<B, BD, 0, stream>>>(
        node_repr, w1h, b1, w2, b2, gumbel, sub_ids, offs, counts, out_idx, out_samples);
}
